// DyRMLP_57612691309384
// MI455X (gfx1250) — matrix operand order repaired and checked by reading, not compiled, not run on hardware
//
#include <hip/hip_runtime.h>
#include <math.h>

#define NUM_E     50000
#define EMBED_DIM 128
#define HID_DIM   200
#define BATCH     512
#define NHIST     50
#define KPAD      224     // HID_DIM padded to 7 chunks of 32 for WMMA K
#define ALPHA     0.5f

typedef _Float16 v16h __attribute__((ext_vector_type(16)));
typedef _Float16 v8h  __attribute__((ext_vector_type(8)));
typedef float    v8f  __attribute__((ext_vector_type(8)));

// ---------------------------------------------------------------------------
// Kernel 1: per-batch-row capsule routing. One 128-thread block per b.
// Produces poses[b,200], c[b,50], pval[b,50] (= p_raw at idx[b,n]) and
// denomP[b] = sum_e exp(p_raw[b,e]) computed analytically from the <=50
// nonzero entries (with duplicate-id multiplicity handled).
// ---------------------------------------------------------------------------
__global__ __launch_bounds__(128)
void routing_kernel(const int* __restrict__ idx, const int* __restrict__ times,
                    const float* __restrict__ emb, const float* __restrict__ Ws_w,
                    const float* __restrict__ Ws_b,
                    float* __restrict__ poses, float* __restrict__ cmat,
                    float* __restrict__ pval, float* __restrict__ denomP)
{
    __shared__ float u[NHIST * HID_DIM];      // u_hat, 40 KB
    __shared__ float blog[NHIST], cvec[NHIST], vvec[HID_DIM];
    __shared__ float term[NHIST], rnorm[NHIST], sscale;
    __shared__ int   sidx[NHIST];

    const int b = blockIdx.x, t = threadIdx.x, T = 128;

    for (int n = t; n < NHIST; n += T) {
        sidx[n] = idx[b * NHIST + n];
        blog[n] = 2.0f / (1.0f + (float)times[b * NHIST + n]);  // (W+1)/(W+t)
    }
    __syncthreads();

    // x[n,h] = embeds[n,:] . Ws_w[h,:] + Ws_b[h]
    for (int p = t; p < NHIST * HID_DIM; p += T) {
        int n = p / HID_DIM, h = p - n * HID_DIM;
        const float* er = emb  + (long long)sidx[n] * EMBED_DIM;
        const float* wr = Ws_w + (long long)h * EMBED_DIM;
        float a = Ws_b[h];
#pragma unroll 8
        for (int d = 0; d < EMBED_DIM; ++d) a = fmaf(er[d], wr[d], a);
        u[p] = a;
    }
    __syncthreads();

    // L2 normalize each of the 50 rows over H
    if (t < NHIST) {
        const float* r = u + t * HID_DIM;
        float s = 0.f;
        for (int h = 0; h < HID_DIM; ++h) s = fmaf(r[h], r[h], s);
        rnorm[t] = 1.0f / fmaxf(sqrtf(s), 1e-12f);
    }
    __syncthreads();
    for (int p = t; p < NHIST * HID_DIM; p += T) u[p] *= rnorm[p / HID_DIM];
    __syncthreads();

    // dynamic routing, 3 iterations
    for (int it = 0; it < 3; ++it) {
        if (t == 0) {   // softmax over 50 logits, scaled by N
            float m = blog[0];
            for (int n = 1; n < NHIST; ++n) m = fmaxf(m, blog[n]);
            float s = 0.f;
            for (int n = 0; n < NHIST; ++n) { cvec[n] = expf(blog[n] - m); s += cvec[n]; }
            float inv = (float)NHIST / s;
            for (int n = 0; n < NHIST; ++n) cvec[n] *= inv;
        }
        __syncthreads();
        for (int h = t; h < HID_DIM; h += T) {   // s = c . u_hat
            float sh = 0.f;
            for (int n = 0; n < NHIST; ++n) sh = fmaf(cvec[n], u[n * HID_DIM + h], sh);
            vvec[h] = sh;
        }
        __syncthreads();
        if (t == 0) {   // squash scale
            float sq = 0.f;
            for (int h = 0; h < HID_DIM; ++h) sq = fmaf(vvec[h], vvec[h], sq);
            sscale = sq / (1.0f + sq) / sqrtf(sq + 1e-9f);
        }
        __syncthreads();
        for (int h = t; h < HID_DIM; h += T) vvec[h] *= sscale;
        __syncthreads();
        if (it < 2 && t < NHIST) {   // b += u_hat . v
            const float* r = u + t * HID_DIM;
            float d = 0.f;
            for (int h = 0; h < HID_DIM; ++h) d = fmaf(r[h], vvec[h], d);
            blog[t] += d;
        }
        __syncthreads();
    }

    for (int h = t; h < HID_DIM; h += T) poses[b * HID_DIM + h] = vvec[h];
    if (t < NHIST) {
        cmat[b * NHIST + t] = cvec[t];
        float pe = 0.f; int mult = 0;
        for (int m = 0; m < NHIST; ++m) {
            bool eq = (sidx[m] == sidx[t]);
            pe += eq ? cvec[m] : 0.f;
            mult += eq ? 1 : 0;
        }
        pval[b * NHIST + t] = pe;
        term[t] = (expf(pe) - 1.0f) / (float)mult;   // distinct-id correction
    }
    __syncthreads();
    if (t == 0) {
        float tot = (float)NUM_E;                    // zeros contribute exp(0)=1 each
        for (int n = 0; n < NHIST; ++n) tot += term[n];
        denomP[b] = tot;
    }
}

// ---------------------------------------------------------------------------
// Kernel 2: f32 -> f16 conversion with K padding 200 -> 224 (zeros in pad).
// ---------------------------------------------------------------------------
__global__ __launch_bounds__(256)
void conv_half_kernel(const float* __restrict__ src, _Float16* __restrict__ dst,
                      long long total)
{
    long long i = (long long)blockIdx.x * 256 + threadIdx.x;
    if (i >= total) return;
    long long row = i / KPAD;
    int k = (int)(i - row * KPAD);
    dst[i] = (k < HID_DIM) ? (_Float16)src[row * HID_DIM + k] : (_Float16)0.f;
}

// ---------------------------------------------------------------------------
// Kernel 3: scores = poses @ mlp_w^T + mlp_b via v_wmma_f32_16x16x32_f16.
// One wave owns a 16-wide e-tile and 4 m-tiles (64 batch rows); B fragment
// reused 4x per K-chunk. 50000/16 = 3125 e-tiles, 8 m-groups -> 25000 waves.
// Fragment layouts follow the ISA 16-bit A (16x32) / B (32x16) VGPR tables.
// ---------------------------------------------------------------------------
__global__ __launch_bounds__(256)
void gemm_kernel(const _Float16* __restrict__ A,     // [BATCH][KPAD] f16
                 const _Float16* __restrict__ W,     // [NUM_E][KPAD] f16
                 const float* __restrict__ bias,     // [NUM_E]
                 float* __restrict__ out)            // [BATCH][NUM_E]
{
    const int wave  = (blockIdx.x * 256 + threadIdx.x) >> 5;
    const int lane  = threadIdx.x & 31;
    const int eTile = wave >> 3;
    const int mGrp  = wave & 7;
    if (eTile >= NUM_E / 16) return;

    const int col = lane & 15;        // N (B/C/D) or M (A) index for this lane
    const int hi  = lane >> 4;        // lane-group selector for K halves
    const int e   = eTile * 16 + col;
    const int m0  = mGrp * 64;

    const _Float16* wrow = W + (long long)e * KPAD + hi * 8;

    v8f acc[4] = {};
    for (int kc = 0; kc < KPAD / 32; ++kc) {
        const int k0 = kc * 32;
        if (kc + 1 < KPAD / 32)
            __builtin_prefetch(wrow + k0 + 32, 0, 0);   // global_prefetch_b8

        // B fragment: 32x16, lane holds column `col`; halves 0..7 = K k0 + hi*8 .. +7, halves 8..15 = K k0+16+hi*8 ..
        v8h blo = *(const v8h*)(wrow + k0);
        v8h bhi = *(const v8h*)(wrow + k0 + 16);
        v16h bfrag = __builtin_shufflevector(blo, bhi, 0,1,2,3,4,5,6,7,8,9,10,11,12,13,14,15);

#pragma unroll
        for (int mt = 0; mt < 4; ++mt) {
            // A fragment: 16x32, lane holds row m0+mt*16+col;
            // halves 0..7  = K k0 + hi*8 .. +7, halves 8..15 = K k0+16+hi*8 ..
            const _Float16* arow = A + (long long)(m0 + mt * 16 + col) * KPAD
                                     + k0 + hi * 8;
            v8h lo = *(const v8h*)(arow);
            v8h hv = *(const v8h*)(arow + 16);
            v16h afrag;
#pragma unroll
            for (int i = 0; i < 8; ++i) { afrag[i] = lo[i]; afrag[i + 8] = hv[i]; }

            acc[mt] = __builtin_amdgcn_wmma_f32_16x16x32_f16(
                false, afrag, false, bfrag, (short)0, acc[mt], false, false);
        }
    }

    const float be = bias[e];
#pragma unroll
    for (int mt = 0; mt < 4; ++mt) {
        const int rbase = m0 + mt * 16 + hi * 8;   // D: VGPR r -> row rbase+r
#pragma unroll
        for (int r = 0; r < 8; ++r)
            out[(long long)(rbase + r) * NUM_E + e] = acc[mt][r] + be;
    }
}

// ---------------------------------------------------------------------------
// Kernel 4: per-row max and sum(exp) over the 50000-wide score rows.
// ---------------------------------------------------------------------------
__global__ __launch_bounds__(256)
void rowstat_kernel(const float* __restrict__ out,
                    float* __restrict__ rowMax, float* __restrict__ rowSum)
{
    __shared__ float red[256];
    const int b = blockIdx.x, t = threadIdx.x;
    const float* row = out + (long long)b * NUM_E;

    float m = -3.4e38f;
    for (int e = t; e < NUM_E; e += 256) m = fmaxf(m, row[e]);
    red[t] = m; __syncthreads();
    for (int s = 128; s > 0; s >>= 1) { if (t < s) red[t] = fmaxf(red[t], red[t + s]); __syncthreads(); }
    const float M = red[0];
    __syncthreads();

    float sum = 0.f;
    for (int e = t; e < NUM_E; e += 256) sum += expf(row[e] - M);
    red[t] = sum; __syncthreads();
    for (int s = 128; s > 0; s >>= 1) { if (t < s) red[t] += red[t + s]; __syncthreads(); }
    if (t == 0) { rowMax[b] = M; rowSum[b] = red[0]; }
}

// ---------------------------------------------------------------------------
// Kernel 5: exact output values at the <=50 gathered columns per row,
// computed BEFORE finalize overwrites the raw scores.
// ---------------------------------------------------------------------------
__global__ __launch_bounds__(256)
void fixcalc_kernel(const int* __restrict__ idx, const float* __restrict__ out,
                    const float* __restrict__ pval, const float* __restrict__ denomP,
                    const float* __restrict__ rowMax, const float* __restrict__ rowSum,
                    float* __restrict__ fix)
{
    int t = blockIdx.x * 256 + threadIdx.x;
    if (t >= BATCH * NHIST) return;
    int b = t / NHIST;
    int e = idx[t];
    float sc   = out[(long long)b * NUM_E + e];
    float sims = expf(sc - rowMax[b]) / rowSum[b];
    float pv   = expf(pval[t]) / denomP[b];
    fix[t] = logf(ALPHA * pv + (1.f - ALPHA) * sims);
}

// ---------------------------------------------------------------------------
// Kernel 6: in-place epilogue for all elements assuming p_raw = 0.
// ---------------------------------------------------------------------------
__global__ __launch_bounds__(256)
void finalize_kernel(float* __restrict__ out, const float* __restrict__ denomP,
                     const float* __restrict__ rowMax, const float* __restrict__ rowSum)
{
    long long i = (long long)blockIdx.x * 256 + threadIdx.x;
    if (i >= (long long)BATCH * NUM_E) return;
    long long b = i / NUM_E;
    float sims = expf(out[i] - rowMax[b]) / rowSum[b];
    out[i] = logf(ALPHA / denomP[b] + (1.f - ALPHA) * sims);
}

// ---------------------------------------------------------------------------
// Kernel 7: scatter the exact values back. Duplicate ids write identical
// values (pval already sums over all duplicates) -> benign, deterministic.
// ---------------------------------------------------------------------------
__global__ __launch_bounds__(256)
void fixwrite_kernel(const int* __restrict__ idx, const float* __restrict__ fix,
                     float* __restrict__ out)
{
    int t = blockIdx.x * 256 + threadIdx.x;
    if (t >= BATCH * NHIST) return;
    int b = t / NHIST;
    out[(long long)b * NUM_E + idx[t]] = fix[t];
}

// ---------------------------------------------------------------------------
extern "C" void kernel_launch(void* const* d_in, const int* in_sizes, int n_in,
                              void* d_out, int out_size, void* d_ws, size_t ws_size,
                              hipStream_t stream)
{
    const int*   idx   = (const int*)  d_in[0];   // [512,50]
    const int*   times = (const int*)  d_in[1];   // [512,50]
    const float* emb   = (const float*)d_in[2];   // [50000,128]
    const float* Ws_w  = (const float*)d_in[3];   // [200,128]
    const float* Ws_b  = (const float*)d_in[4];   // [200]
    const float* mlp_w = (const float*)d_in[5];   // [50000,200]
    const float* mlp_b = (const float*)d_in[6];   // [50000]
    float* out = (float*)d_out;                   // [512,50000]

    // workspace carve-up (256B aligned)
    auto align256 = [](size_t x) { return (x + 255) & ~(size_t)255; };
    char* ws = (char*)d_ws;
    float* poses  = (float*)ws;  ws += align256((size_t)BATCH * HID_DIM * 4);
    float* cmat   = (float*)ws;  ws += align256((size_t)BATCH * NHIST * 4);
    float* pval   = (float*)ws;  ws += align256((size_t)BATCH * NHIST * 4);
    float* denomP = (float*)ws;  ws += align256((size_t)BATCH * 4);
    float* rowMax = (float*)ws;  ws += align256((size_t)BATCH * 4);
    float* rowSum = (float*)ws;  ws += align256((size_t)BATCH * 4);
    float* fix    = (float*)ws;  ws += align256((size_t)BATCH * NHIST * 4);
    _Float16* Ah  = (_Float16*)ws; ws += align256((size_t)BATCH * KPAD * 2);
    _Float16* Wh  = (_Float16*)ws; ws += align256((size_t)NUM_E * KPAD * 2);

    // 1) routing -> poses, c, pval, denomP
    routing_kernel<<<BATCH, 128, 0, stream>>>(idx, times, emb, Ws_w, Ws_b,
                                              poses, cmat, pval, denomP);
    // 2) f16 conversions with K padding
    {
        long long tA = (long long)BATCH * KPAD;
        long long tW = (long long)NUM_E * KPAD;
        conv_half_kernel<<<(int)((tA + 255) / 256), 256, 0, stream>>>(poses, Ah, tA);
        conv_half_kernel<<<(int)((tW + 255) / 256), 256, 0, stream>>>(mlp_w, Wh, tW);
    }
    // 3) WMMA GEMM -> raw scores in d_out
    {
        int waves  = (NUM_E / 16) * 8;          // 25000
        int blocks = (waves * 32 + 255) / 256;  // 3125
        gemm_kernel<<<blocks, 256, 0, stream>>>(Ah, Wh, mlp_b, out);
    }
    // 4) row softmax stats
    rowstat_kernel<<<BATCH, 256, 0, stream>>>(out, rowMax, rowSum);
    // 5) exact values at gathered columns (before overwrite)
    fixcalc_kernel<<<(BATCH * NHIST + 255) / 256, 256, 0, stream>>>(
        idx, out, pval, denomP, rowMax, rowSum, fix);
    // 6) in-place epilogue everywhere (p_raw = 0 case)
    {
        long long total = (long long)BATCH * NUM_E;
        finalize_kernel<<<(int)((total + 255) / 256), 256, 0, stream>>>(
            out, denomP, rowMax, rowSum);
    }
    // 7) scatter exact values back
    fixwrite_kernel<<<(BATCH * NHIST + 255) / 256, 256, 0, stream>>>(idx, fix, out);
}
